// FAFMoudle_74775380623549
// MI455X (gfx1250) — matrix operand order repaired and checked by reading, not compiled, not run on hardware
//
#include <hip/hip_runtime.h>
#include <math.h>

typedef __attribute__((ext_vector_type(16))) _Float16 v16h;
typedef __attribute__((ext_vector_type(8)))  _Float16 v8h;
typedef __attribute__((ext_vector_type(8)))  float    v8f;

#define HW    81
#define BATCH 256
#define CHI   768
#define CLO   64

__device__ __forceinline__ float wave_sum(float v) {
#pragma unroll
  for (int off = 16; off > 0; off >>= 1) v += __shfl_xor(v, off, 32);
  return v;
}

// A fragment: two contiguous 8-half runs (ISA K-map: k = hi*8+{0..7}, 16+hi*8+{0..7})
__device__ __forceinline__ v16h load_a16(const _Float16* p) {
  v8h lo  = *(const v8h*)p;
  v8h hi8 = *(const v8h*)(p + 16);
  v16h a;
#pragma unroll
  for (int h = 0; h < 8; ++h) { a[h] = lo[h]; a[h + 8] = hi8[h]; }
  return a;
}

// ---------------------------------------------------------------------------
// Weight prep: fp32 -> f16 (same layout)
// ---------------------------------------------------------------------------
__global__ void k_cvt_f16(const float* __restrict__ src, _Float16* __restrict__ dst, int n) {
  int i = blockIdx.x * blockDim.x + threadIdx.x;
  if (i < n) dst[i] = (_Float16)src[i];
}

// w_bconv (O,C,3,3) fp32 -> f16 [tap][o][c] so conv A-rows are contiguous in c
__global__ void k_cvt_bconv(const float* __restrict__ w, _Float16* __restrict__ dst) {
  int idx = blockIdx.x * blockDim.x + threadIdx.x;  // over 768*768*9
  if (idx >= CHI * CHI * 9) return;
  int o = idx / (CHI * 9);
  int r = idx % (CHI * 9);
  int c = r / 9, t = r % 9;
  dst[((size_t)t * CHI + o) * CHI + c] = (_Float16)w[idx];
}

// ---------------------------------------------------------------------------
// Batched 1x1-conv GEMM via WMMA f16 (f32 accumulate).
// Each wave owns a 64-row o-strip x 16-pixel tile: per K-step one shared
// B-fragment feeds 4 back-to-back WMMAs (6 loads/WMMA). All loads are
// unconditional (edge columns clamp to pixel 80; results unstored).
// grid: (6 pixel-tiles, Odim/256, BATCH); block 128 (4 waves x 64 rows).
// ---------------------------------------------------------------------------
__global__ void k_gemm1x1(const _Float16* __restrict__ Wt, const float* __restrict__ bias,
                          const float* __restrict__ X, float* __restrict__ Out,
                          int Cdim, int outStrideC, int chOff) {
  const int lane = threadIdx.x & 31;
  const int wv   = threadIdx.x >> 5;
  const int p0   = blockIdx.x * 16;
  const int o0   = (blockIdx.y * 4 + wv) * 64;
  const int b    = blockIdx.z;
  const int m    = lane & 15;
  const int hi   = lane >> 4;
  const int p    = p0 + m;
  const bool pin = p < HW;
  const int pc   = pin ? p : (HW - 1);   // clamped: always-valid address

  v8f acc[4] = {};
  const _Float16* wbase = Wt + (size_t)(o0 + m) * Cdim + hi * 8;
  const float*    xbase = X + (size_t)b * Cdim * HW + pc;

  for (int k0 = 0; k0 < Cdim; k0 += 32) {
    const float* xc = xbase + (size_t)(k0 + hi * 8) * HW;
    v16h bf;
#pragma unroll
    for (int h = 0; h < 16; ++h) bf[h] = (_Float16)xc[(size_t)(h + (h & 8)) * HW];
    if (k0 + 32 < Cdim)
      __builtin_prefetch(xbase + (size_t)(k0 + 32) * HW, 0, 3);  // global_prefetch_b8

    v16h a0 = load_a16(wbase + (size_t)0  * Cdim + k0);
    v16h a1 = load_a16(wbase + (size_t)16 * Cdim + k0);
    v16h a2 = load_a16(wbase + (size_t)32 * Cdim + k0);
    v16h a3 = load_a16(wbase + (size_t)48 * Cdim + k0);
    acc[0] = __builtin_amdgcn_wmma_f32_16x16x32_f16(false, a0, false, bf, (short)0, acc[0], false, false);
    acc[1] = __builtin_amdgcn_wmma_f32_16x16x32_f16(false, a1, false, bf, (short)0, acc[1], false, false);
    acc[2] = __builtin_amdgcn_wmma_f32_16x16x32_f16(false, a2, false, bf, (short)0, acc[2], false, false);
    acc[3] = __builtin_amdgcn_wmma_f32_16x16x32_f16(false, a3, false, bf, (short)0, acc[3], false, false);
  }

  if (pin) {
#pragma unroll
    for (int j = 0; j < 4; ++j)
#pragma unroll
      for (int r = 0; r < 8; ++r) {
        int o = o0 + j * 16 + hi * 8 + r;            // ISA C/D layout
        Out[((size_t)b * outStrideC + chOff + o) * HW + p] = acc[j][r] + bias[o];
      }
  }
}

// ---------------------------------------------------------------------------
// 3x3 SAME conv (no bias) + BatchNorm + LeakyReLU via WMMA.
// Weights pre-laid-out f16 [tap][o][c]; 9 shifted K-panels; halo zeroed by
// VALU multiply on clamped unconditional loads. Same 4-tile B-reuse scheme.
// grid: (6, 3, BATCH); block 128.
// ---------------------------------------------------------------------------
__global__ void k_conv3x3_bn(const _Float16* __restrict__ Wtr,  // [9][768][768] f16
                             const float* __restrict__ Xin,     // (B, 768, 9, 9)
                             const float* __restrict__ gamma, const float* __restrict__ beta,
                             const float* __restrict__ mean,  const float* __restrict__ var,
                             float* __restrict__ Out) {
  const int lane = threadIdx.x & 31;
  const int wv   = threadIdx.x >> 5;
  const int p0   = blockIdx.x * 16;
  const int o0   = (blockIdx.y * 4 + wv) * 64;
  const int b    = blockIdx.z;
  const int m    = lane & 15;
  const int hi   = lane >> 4;
  const int p    = p0 + m;
  const bool pin = p < HW;
  const int pc   = pin ? p : (HW - 1);
  const int h9 = pc / 9, w9 = pc % 9;

  v8f acc[4] = {};
  const float* xb0 = Xin + (size_t)b * CHI * HW;

  for (int t = 0; t < 9; ++t) {
    const int dh = t / 3 - 1, dw = t % 3 - 1;
    const int ih = h9 + dh, iw = w9 + dw;
    const bool ok = ((unsigned)ih < 9u) && ((unsigned)iw < 9u);
    const float okf = ok ? 1.f : 0.f;            // halo zero via VALU multiply
    const int sofs = ok ? (ih * 9 + iw) : pc;    // always-valid address
    const _Float16* wbase = Wtr + ((size_t)t * CHI + o0 + m) * CHI + hi * 8;
    const float* xb = xb0 + sofs;
    for (int k0 = 0; k0 < CHI; k0 += 32) {
      const float* xc = xb + (size_t)(k0 + hi * 8) * HW;
      v16h bf;
#pragma unroll
      for (int h = 0; h < 16; ++h) bf[h] = (_Float16)(xc[(size_t)(h + (h & 8)) * HW] * okf);

      v16h a0 = load_a16(wbase + (size_t)0  * CHI + k0);
      v16h a1 = load_a16(wbase + (size_t)16 * CHI + k0);
      v16h a2 = load_a16(wbase + (size_t)32 * CHI + k0);
      v16h a3 = load_a16(wbase + (size_t)48 * CHI + k0);
      acc[0] = __builtin_amdgcn_wmma_f32_16x16x32_f16(false, a0, false, bf, (short)0, acc[0], false, false);
      acc[1] = __builtin_amdgcn_wmma_f32_16x16x32_f16(false, a1, false, bf, (short)0, acc[1], false, false);
      acc[2] = __builtin_amdgcn_wmma_f32_16x16x32_f16(false, a2, false, bf, (short)0, acc[2], false, false);
      acc[3] = __builtin_amdgcn_wmma_f32_16x16x32_f16(false, a3, false, bf, (short)0, acc[3], false, false);
    }
  }

  if (pin) {
#pragma unroll
    for (int j = 0; j < 4; ++j)
#pragma unroll
      for (int r = 0; r < 8; ++r) {
        int o = o0 + j * 16 + hi * 8 + r;
        float sc = gamma[o] * rsqrtf(var[o] + 1e-5f);
        float v  = (acc[j][r] - mean[o]) * sc + beta[o];
        v = v > 0.f ? v : 0.01f * v;             // leaky_relu 0.01
        Out[((size_t)b * CHI + o) * HW + p] = v;
      }
  }
}

// ---------------------------------------------------------------------------
// fuse_3 = conv1_2(concat(convh3(x), convl2(y))), fuse_4 = convl3(y).
// ---------------------------------------------------------------------------
__global__ void k_small_fuses(const float* __restrict__ x, const float* __restrict__ y,
                              const float* __restrict__ w_h3, const float* __restrict__ b_h3,
                              const float* __restrict__ w_l2, const float* __restrict__ b_l2,
                              const float* __restrict__ w_c12, const float* __restrict__ b_c12,
                              const float* __restrict__ w_l3, const float* __restrict__ b_l3,
                              float* __restrict__ fuse3, float* __restrict__ fuse4) {
  const int lane = threadIdx.x & 31;
  const int pix  = blockIdx.x * (blockDim.x >> 5) + (threadIdx.x >> 5);
  if (pix >= BATCH * HW) return;
  const int b = pix / HW, hw = pix % HW;
  const float* xb = x + (size_t)b * CHI * HW + hw;
  const float* yb = y + (size_t)b * CLO * HW + hw;

  float pa = 0.f, pb = 0.f, p40 = 0.f, p41 = 0.f;
  for (int c = lane; c < CHI; c += 32) pa += w_h3[c] * xb[(size_t)c * HW];
  for (int c = lane; c < CLO; c += 32) {
    float yv = yb[(size_t)c * HW];
    pb  += w_l2[c] * yv;
    p40 += w_l3[c] * yv;
    p41 += w_l3[CLO + c] * yv;
  }
  pa = wave_sum(pa); pb = wave_sum(pb);
  p40 = wave_sum(p40); p41 = wave_sum(p41);
  if (lane == 0) {
    float t3a = pa + b_h3[0], t3b = pb + b_l2[0];
    size_t base = (size_t)b * 2 * HW + hw;
    fuse3[base]      = w_c12[0] * t3a + w_c12[1] * t3b + b_c12[0];
    fuse3[base + HW] = w_c12[2] * t3a + w_c12[3] * t3b + b_c12[1];
    fuse4[base]      = p40 + b_l3[0];
    fuse4[base + HW] = p41 + b_l3[1];
  }
}

// stack[(q*2+ch)*20736 + b*81 + hw] for q in {x, y, x^2, y^2, x*y}
__global__ void k_ssim_prep(const float* __restrict__ f3, const float* __restrict__ f4,
                            float* __restrict__ stack) {
  int idx = blockIdx.x * blockDim.x + threadIdx.x;
  if (idx >= 2 * BATCH * HW) return;
  int b = idx / (2 * HW);
  int r = idx % (2 * HW);
  int ch = r / HW, hw = r % HW;
  float a = f3[idx], c = f4[idx];
  const size_t QC = (size_t)BATCH * HW;
  size_t pos = (size_t)ch * QC + (size_t)b * HW + hw;
  stack[pos]          = a;
  stack[pos + 2 * QC] = c;
  stack[pos + 4 * QC] = a * a;
  stack[pos + 6 * QC] = c * c;
  stack[pos + 8 * QC] = a * c;
}

// uniform_filter1d, window 7, mode='reflect' (symmetric), along middle dim.
__global__ void k_ufilt(const float* __restrict__ in, float* __restrict__ out,
                        int N, int innerStride, int total) {
  int idx = blockIdx.x * blockDim.x + threadIdx.x;
  if (idx >= total) return;
  int inner = idx % innerStride;
  int rest  = idx / innerStride;
  int i     = rest % N;
  int outer = rest / N;
  const float* basep = in + (size_t)outer * N * innerStride + inner;
  float s = 0.f;
#pragma unroll
  for (int t = -3; t <= 3; ++t) {
    int j = i + t;
    if (j < 0)  j = -1 - j;
    if (j >= N) j = 2 * N - 1 - j;
    s += basep[(size_t)j * innerStride];
  }
  out[idx] = s * (1.f / 7.f);
}

__global__ void k_ssim_map(const float* __restrict__ F, float* __restrict__ ssim) {
  int idx = blockIdx.x * blockDim.x + threadIdx.x;
  if (idx >= 2 * BATCH * HW) return;
  int b = idx / (2 * HW);
  int r = idx % (2 * HW);
  int ch = r / HW, hw = r % HW;
  const size_t QC = (size_t)BATCH * HW;
  size_t base = (size_t)ch * QC + (size_t)b * HW + hw;
  float ux  = F[base];
  float uy  = F[base + 2 * QC];
  float uxx = F[base + 4 * QC];
  float uyy = F[base + 6 * QC];
  float uxy = F[base + 8 * QC];
  const float cov = 343.f / 342.f, c1 = 1e-4f, c2 = 9e-4f;
  float vx  = cov * (uxx - ux * ux);
  float vy  = cov * (uyy - uy * uy);
  float vxy = cov * (uxy - ux * uy);
  ssim[idx] = ((2.f * ux * uy + c1) * (2.f * vxy + c2)) /
              ((ux * ux + uy * uy + c1) * (vx + vy + c2));
}

// cor_1 = cossim(fuse_1, fuse_2); fuse2_1 = fuse_1 + 0.5*(1-cor_1)*fuse_2 (in-place)
__global__ void k_cor1_fuse21(float* __restrict__ f1, const float* __restrict__ f2) {
  const int lane = threadIdx.x & 31;
  const int pix  = blockIdx.x * (blockDim.x >> 5) + (threadIdx.x >> 5);
  if (pix >= BATCH * HW) return;
  const int b = pix / HW, hw = pix % HW;
  float* ab = f1 + (size_t)b * CHI * HW + hw;
  const float* bb = f2 + (size_t)b * CHI * HW + hw;
  float dot = 0.f, na = 0.f, nb = 0.f;
  for (int c = lane; c < CHI; c += 32) {
    float av = ab[(size_t)c * HW], bv = bb[(size_t)c * HW];
    dot += av * bv; na += av * av; nb += bv * bv;
  }
  dot = wave_sum(dot); na = wave_sum(na); nb = wave_sum(nb);
  float cor = dot / fmaxf(sqrtf(na) * sqrtf(nb), 1e-8f);
  float sc  = 0.5f * (1.f - cor);
  for (int c = lane; c < CHI; c += 32)
    ab[(size_t)c * HW] += sc * bb[(size_t)c * HW];
}

// fuse2_2 = cc1(fuse_4 + ssim*fuse_3); scc = cc1(ssim)
__global__ void k_fuse22(const float* __restrict__ f3, const float* __restrict__ f4,
                         const float* __restrict__ ssim,
                         const float* __restrict__ w_cc1, const float* __restrict__ b_cc1,
                         float* __restrict__ f22, float* __restrict__ scc) {
  int pix = blockIdx.x * blockDim.x + threadIdx.x;
  if (pix >= BATCH * HW) return;
  int b = pix / HW, hw = pix % HW;
  size_t i0 = (size_t)b * 2 * HW + hw, i1 = i0 + HW;
  float s0 = ssim[i0], s1 = ssim[i1];
  float u0 = f4[i0] + s0 * f3[i0];
  float u1 = f4[i1] + s1 * f3[i1];
  f22[pix] = w_cc1[0] * u0 + w_cc1[1] * u1 + b_cc1[0];
  scc[pix] = w_cc1[0] * s0 + w_cc1[1] * s1 + b_cc1[0];
}

// cor2_1 = cossim(fuse2_1, bcast(fuse2_2)); fuse3_1 += 0.5*(1-cor2+scc)*f22
__global__ void k_cor2_fuse31(float* __restrict__ f21, const float* __restrict__ f22,
                              const float* __restrict__ scc) {
  const int lane = threadIdx.x & 31;
  const int pix  = blockIdx.x * (blockDim.x >> 5) + (threadIdx.x >> 5);
  if (pix >= BATCH * HW) return;
  const int b = pix / HW, hw = pix % HW;
  float* ab = f21 + (size_t)b * CHI * HW + hw;
  float s = 0.f, ss = 0.f;
  for (int c = lane; c < CHI; c += 32) {
    float v = ab[(size_t)c * HW];
    s += v; ss += v * v;
  }
  s = wave_sum(s); ss = wave_sum(ss);
  float w = f22[pix];
  float cor = (w * s) / fmaxf(sqrtf(ss) * fabsf(w) * sqrtf((float)CHI), 1e-8f);
  float t = 0.5f * (1.f - cor + scc[pix]);
  float add = t * w;
  for (int c = lane; c < CHI; c += 32)
    ab[(size_t)c * HW] += add;
}

// xc = concat(f22, conv3x3(f22, w_pool)+b_pool, f22) flattened -> (b, 324)
__global__ void k_xcpool(const float* __restrict__ f22,
                         const float* __restrict__ w_pool, const float* __restrict__ b_pool,
                         float* __restrict__ xc) {
  int pix = blockIdx.x * blockDim.x + threadIdx.x;
  if (pix >= BATCH * HW) return;
  int b = pix / HW, hw = pix % HW;
  int h9 = hw / 9, w9 = hw % 9;
  const float* fb = f22 + (size_t)b * HW;
  float p0 = b_pool[0], p1 = b_pool[1];
#pragma unroll
  for (int kh = 0; kh < 3; ++kh)
#pragma unroll
    for (int kw = 0; kw < 3; ++kw) {
      int ih = h9 + kh - 1, iw = w9 + kw - 1;
      if ((unsigned)ih < 9u && (unsigned)iw < 9u) {
        float v = fb[ih * 9 + iw];
        p0 += w_pool[kh * 3 + kw] * v;
        p1 += w_pool[9 + kh * 3 + kw] * v;
      }
    }
  float* xb = xc + (size_t)b * 324;
  float fv = fb[hw];
  xb[hw] = fv; xb[81 + hw] = p0; xb[162 + hw] = p1; xb[243 + hw] = fv;
}

__global__ void k_fc1(const float* __restrict__ xc, const float* __restrict__ w,
                      const float* __restrict__ bias, float* __restrict__ h1) {
  int idx = blockIdx.x * blockDim.x + threadIdx.x;
  if (idx >= BATCH * 324) return;
  int b = idx / 324, j = idx % 324;
  const float* xb = xc + (size_t)b * 324;
  const float* wr = w + (size_t)j * 324;
  float s = bias[j];
  for (int i = 0; i < 324; ++i) s += xb[i] * wr[i];
  h1[idx] = 0.5f * s * (1.f + erff(s * 0.70710678118654752f));  // exact GELU
}

__global__ void k_fc2(const float* __restrict__ h1, const float* __restrict__ w,
                      const float* __restrict__ bias, float* __restrict__ xw) {
  int idx = blockIdx.x * blockDim.x + threadIdx.x;
  if (idx >= BATCH * HW) return;
  int b = idx / HW, j = idx % HW;
  const float* hb = h1 + (size_t)b * 324;
  const float* wr = w + (size_t)j * 324;
  float s = bias[j];
  for (int i = 0; i < 324; ++i) s += hb[i] * wr[i];
  xw[idx] = s > 0.f ? s : 0.01f * s;
}

// xout = fuse3_1 * (1 + xweight), in-place on bufA
__global__ void k_xout(float* __restrict__ A, const float* __restrict__ xw) {
  size_t idx = (size_t)blockIdx.x * blockDim.x + threadIdx.x;
  if (idx >= (size_t)BATCH * CHI * HW) return;
  int b  = (int)(idx / ((size_t)CHI * HW));
  int hw = (int)(idx % HW);
  A[idx] *= (1.f + xw[b * HW + hw]);
}

// ---------------------------------------------------------------------------
extern "C" void kernel_launch(void* const* d_in, const int* in_sizes, int n_in,
                              void* d_out, int out_size, void* d_ws, size_t ws_size,
                              hipStream_t stream) {
  const float* x        = (const float*)d_in[0];
  const float* y        = (const float*)d_in[1];
  const float* w_convh1 = (const float*)d_in[2];
  const float* b_convh1 = (const float*)d_in[3];
  const float* w_conv11 = (const float*)d_in[4];
  const float* b_conv11 = (const float*)d_in[5];
  const float* w_convh2 = (const float*)d_in[6];
  const float* b_convh2 = (const float*)d_in[7];
  const float* w_convl1 = (const float*)d_in[8];
  const float* b_convl1 = (const float*)d_in[9];
  const float* w_convh3 = (const float*)d_in[10];
  const float* b_convh3 = (const float*)d_in[11];
  const float* w_convl2 = (const float*)d_in[12];
  const float* b_convl2 = (const float*)d_in[13];
  const float* w_conv12 = (const float*)d_in[14];
  const float* b_conv12 = (const float*)d_in[15];
  const float* w_convl3 = (const float*)d_in[16];
  const float* b_convl3 = (const float*)d_in[17];
  const float* w_cc1    = (const float*)d_in[18];
  const float* b_cc1    = (const float*)d_in[19];
  const float* w_pool   = (const float*)d_in[20];
  const float* b_pool   = (const float*)d_in[21];
  const float* w_fc1    = (const float*)d_in[22];
  const float* b_fc1    = (const float*)d_in[23];
  const float* w_fc2    = (const float*)d_in[24];
  const float* b_fc2    = (const float*)d_in[25];
  const float* w_bconv  = (const float*)d_in[26];
  const float* bn_gamma = (const float*)d_in[27];
  const float* bn_beta  = (const float*)d_in[28];
  const float* bn_mean  = (const float*)d_in[29];
  const float* bn_var   = (const float*)d_in[30];

  char* ws = (char*)d_ws;
  size_t off = 0;
  auto carve = [&](size_t bytes) -> void* {
    void* p = ws + off;
    off += (bytes + 255) & ~(size_t)255;
    return p;
  };
  const size_t S1  = (size_t)BATCH * CHI * HW * sizeof(float);  // 63.7 MB
  const size_t SZ2 = (size_t)BATCH * 2 * HW * sizeof(float);    // 165888 B

  float* bufA  = (float*)carve(S1);        // fuse_1 -> fuse2_1 -> fuse3_1 -> xout
  float* bufB  = (float*)carve(S1);        // concat(convh2(x), convl1(y))
  float* bufC  = (float*)carve(S1);        // fuse_2
  float* f3    = (float*)carve(SZ2);
  float* f4    = (float*)carve(SZ2);
  float* ssim  = (float*)carve(SZ2);
  float* st0   = (float*)carve(5 * SZ2);
  float* st1   = (float*)carve(5 * SZ2);
  float* f22   = (float*)carve((size_t)BATCH * HW * sizeof(float));
  float* scc   = (float*)carve((size_t)BATCH * HW * sizeof(float));
  float* xc    = (float*)carve((size_t)BATCH * 324 * sizeof(float));
  float* h1b   = (float*)carve((size_t)BATCH * 324 * sizeof(float));
  float* xw    = (float*)carve((size_t)BATCH * HW * sizeof(float));
  _Float16* whc1  = (_Float16*)carve((size_t)CHI * CHI * 2);
  _Float16* whc11 = (_Float16*)carve((size_t)CHI * CHI * 2);
  _Float16* whc2  = (_Float16*)carve((size_t)512 * CHI * 2);
  _Float16* whl1  = (_Float16*)carve((size_t)256 * CLO * 2);
  _Float16* wbch  = (_Float16*)carve((size_t)9 * CHI * CHI * 2);
  (void)ws_size; (void)in_sizes; (void)n_in; (void)out_size;

  const dim3 blk(128);
  const int TOT = 5 * 2 * BATCH * HW;   // 207360 stacked filter elems

  // Weight prep (fp32 -> f16, contiguous A-fragment layouts)
  k_cvt_f16<<<(CHI * CHI + 255) / 256, 256, 0, stream>>>(w_convh1, whc1, CHI * CHI);
  k_cvt_f16<<<(CHI * CHI + 255) / 256, 256, 0, stream>>>(w_conv11, whc11, CHI * CHI);
  k_cvt_f16<<<(512 * CHI + 255) / 256, 256, 0, stream>>>(w_convh2, whc2, 512 * CHI);
  k_cvt_f16<<<(256 * CLO + 255) / 256, 256, 0, stream>>>(w_convl1, whl1, 256 * CLO);
  k_cvt_bconv<<<(CHI * CHI * 9 + 255) / 256, 256, 0, stream>>>(w_bconv, wbch);

  // Big 1x1 GEMMs (WMMA f16, f32 accumulate; 64-row strip per wave)
  k_gemm1x1<<<dim3(6, 3, BATCH), blk, 0, stream>>>(whc1,  b_convh1, x, bufA, CHI, CHI, 0);
  k_gemm1x1<<<dim3(6, 2, BATCH), blk, 0, stream>>>(whc2,  b_convh2, x, bufB, CHI, CHI, 0);
  k_gemm1x1<<<dim3(6, 1, BATCH), blk, 0, stream>>>(whl1,  b_convl1, y, bufB, CLO, CHI, 512);
  k_gemm1x1<<<dim3(6, 3, BATCH), blk, 0, stream>>>(whc11, b_conv11, bufB, bufC, CHI, CHI, 0);

  // Small fuse branches + SSIM (batch/h/w separable uniform filters)
  k_small_fuses<<<5184, 128, 0, stream>>>(x, y, w_convh3, b_convh3, w_convl2, b_convl2,
                                          w_conv12, b_conv12, w_convl3, b_convl3, f3, f4);
  k_ssim_prep<<<(2 * BATCH * HW + 255) / 256, 256, 0, stream>>>(f3, f4, st0);
  k_ufilt<<<(TOT + 255) / 256, 256, 0, stream>>>(st0, st1, BATCH, HW, TOT);  // axis b
  k_ufilt<<<(TOT + 255) / 256, 256, 0, stream>>>(st1, st0, 9, 9, TOT);       // axis h
  k_ufilt<<<(TOT + 255) / 256, 256, 0, stream>>>(st0, st1, 9, 1, TOT);       // axis w
  k_ssim_map<<<(2 * BATCH * HW + 255) / 256, 256, 0, stream>>>(st1, ssim);

  // Cosine-sim fusions
  k_cor1_fuse21<<<5184, 128, 0, stream>>>(bufA, bufC);
  k_fuse22<<<(BATCH * HW + 255) / 256, 256, 0, stream>>>(f3, f4, ssim, w_cc1, b_cc1, f22, scc);
  k_cor2_fuse31<<<5184, 128, 0, stream>>>(bufA, f22, scc);

  // Attention-weight MLP
  k_xcpool<<<(BATCH * HW + 255) / 256, 256, 0, stream>>>(f22, w_pool, b_pool, xc);
  k_fc1<<<(BATCH * 324 + 255) / 256, 256, 0, stream>>>(xc, w_fc1, b_fc1, h1b);
  k_fc2<<<(BATCH * HW + 255) / 256, 256, 0, stream>>>(h1b, w_fc2, b_fc2, xw);
  k_xout<<<(int)(((size_t)BATCH * CHI * HW + 255) / 256), 256, 0, stream>>>(bufA, xw);

  // Final 3x3 conv + BN + leaky (WMMA, BN fused in epilogue)
  k_conv3x3_bn<<<dim3(6, 3, BATCH), blk, 0, stream>>>(wbch, bufA, bn_gamma, bn_beta,
                                                      bn_mean, bn_var, (float*)d_out);
}
